// BertSelfAttention_12618613915827
// MI455X (gfx1250) — compile-verified, matrix operand order checked by reading, not run on hardware
//
#include <hip/hip_runtime.h>

// BERT self-attention forward for MI455X (gfx1250, wave32, WMMA).
// B=4, S=2048, D=1024, H=16, HD=64.
// Pipeline: fp32->bf16 convert, fused QKV GEMM (bf16 WMMA, f32 acc,
// double-buffered GLOBAL_LOAD_ASYNC_TO_LDS staging), flash-attention
// (block-shared async K/V staging, bf16 WMMA QK^T and PV, f32 online softmax).

#define BN 4
#define SN 2048
#define DN 1024
#define HN 16
#define HDN 64
#define MROWS (BN * SN)        // 8192
#define NQKV (3 * DN)          // 3072

typedef __attribute__((ext_vector_type(16))) __bf16 bfrag_t;  // 16 bf16 = 8 VGPRs
typedef __attribute__((ext_vector_type(8)))  float  v8f;      // 16x16 f32 C/D

union FragU { uint4 u[2]; bfrag_t f; };

static __device__ inline bfrag_t make_frag(uint4 a, uint4 b) {
    FragU x; x.u[0] = a; x.u[1] = b; return x.f;
}

// Round-to-nearest-even fp32 -> bf16 (bit pattern as ushort).
static __device__ inline unsigned int f32_to_bf16(float f) {
    unsigned int u = __float_as_uint(f);
    return (u + 0x7FFFu + ((u >> 16) & 1u)) >> 16;
}

// A/B fragment gather from a row-major bf16 row pointer (window of 32 K values).
// ISA 16-bit A layout: lanes 0-15 hold K=0..7 & 16..23, lanes 16-31 hold K=8..15 & 24..31.
static __device__ inline bfrag_t load_frag_row(const unsigned short* p, int hl) {
    uint4 a = *(const uint4*)(p + hl * 8);
    uint4 b = *(const uint4*)(p + 16 + hl * 8);
    return make_frag(a, b);
}

// Async global -> LDS copy, 16B per lane (GLOBAL_LOAD_ASYNC_TO_LDS_B128,
// tracked by ASYNCcnt). Generic LDS pointer's low 32 bits are the LDS offset.
static __device__ inline void async_copy_b128(const unsigned short* gsrc,
                                              unsigned short* ldst) {
    unsigned lds_off = (unsigned)(size_t)ldst;
    unsigned long long gaddr = (unsigned long long)(size_t)gsrc;
    asm volatile("global_load_async_to_lds_b128 %0, %1, off"
                 :: "v"(lds_off), "v"(gaddr) : "memory");
}

// ---------------------------------------------------------------------------
// Kernel 0: fp32 -> bf16 conversion (vectorized 4-wide).
// ---------------------------------------------------------------------------
__global__ void cvt_f32_bf16_kernel(const float* __restrict__ src,
                                    unsigned short* __restrict__ dst, int n4) {
    int i = blockIdx.x * blockDim.x + threadIdx.x;
    if (i < n4) {
        float4 v = ((const float4*)src)[i];
        uint2 o;
        o.x = f32_to_bf16(v.x) | (f32_to_bf16(v.y) << 16);
        o.y = f32_to_bf16(v.z) | (f32_to_bf16(v.w) << 16);
        ((uint2*)dst)[i] = o;
    }
}

// ---------------------------------------------------------------------------
// Kernel 1: fused QKV projection GEMM.
//   Y[m,n] = sum_k X[m,k] * W[n,k] + bias[n]     (M=8192, N=3072, K=1024)
// Block tile 128(M) x 128(N), K-step 32, double-buffered async LDS staging.
// 8 waves as 4(M) x 2(N); each wave computes 32x64 (2x4 WMMA 16x16x32 bf16).
// Q gets 1/sqrt(HD)=0.125 folded in. Outputs per-head:
//   Qb,Kb: [B*H][S][64] bf16 row-major;  Vt: [B*H][64][S] bf16 (transposed).
// ---------------------------------------------------------------------------
__global__ __launch_bounds__(256) void qkv_gemm_kernel(
    const unsigned short* __restrict__ Xb,   // [8192][1024] bf16
    const unsigned short* __restrict__ Wb,   // [3072][1024] bf16 (Wq;Wk;Wv rows)
    const float* __restrict__ bq, const float* __restrict__ bk,
    const float* __restrict__ bv,
    unsigned short* __restrict__ Qb, unsigned short* __restrict__ Kb,
    unsigned short* __restrict__ Vt) {

    __shared__ unsigned short As[2][128 * 40];  // 128 rows x 32 cols, stride 40
    __shared__ unsigned short Bs[2][128 * 40];

    const int tid   = threadIdx.x;
    const int wave  = tid >> 5;
    const int lane  = tid & 31;
    const int ln    = lane & 15;
    const int hl    = lane >> 4;
    const int waveM = wave & 3;   // 4 waves over M (32 rows each)
    const int waveN = wave >> 2;  // 2 waves over N (64 cols each)

    const int mBase = blockIdx.x * 128;
    const int nBase = blockIdx.y * 128;

    v8f acc[2][4] = {};

    // issue async copies of one 128x32 A tile + 128x32 B tile (4 ops/thread)
    auto issue_tile = [&](int bufSel, int kb) {
        #pragma unroll
        for (int t = 0; t < 2; ++t) {
            int c = t * 256 + tid;
            int row = c >> 2, cc = c & 3;
            async_copy_b128(Xb + (size_t)(mBase + row) * DN + kb + cc * 8,
                            &As[bufSel][row * 40 + cc * 8]);
            async_copy_b128(Wb + (size_t)(nBase + row) * DN + kb + cc * 8,
                            &Bs[bufSel][row * 40 + cc * 8]);
        }
    };

    issue_tile(0, 0);
    int buf = 0;
    for (int kb = 0; kb < DN; kb += 32) {
        bool has_next = (kb + 32) < DN;
        if (has_next) {
            issue_tile(buf ^ 1, kb + 32);
            // async ops complete in order: waiting to <=4 drains the tile we
            // are about to consume while the next tile stays in flight.
            asm volatile("s_wait_asynccnt 4" ::: "memory");
        } else {
            asm volatile("s_wait_asynccnt 0" ::: "memory");
        }
        __syncthreads();

        bfrag_t a[2], b[4];
        #pragma unroll
        for (int mi = 0; mi < 2; ++mi) {
            int r = waveM * 32 + mi * 16 + ln;
            uint4 u0 = *(const uint4*)(&As[buf][r * 40 + hl * 8]);
            uint4 u1 = *(const uint4*)(&As[buf][r * 40 + 16 + hl * 8]);
            a[mi] = make_frag(u0, u1);
        }
        #pragma unroll
        for (int ni = 0; ni < 4; ++ni) {
            int r = waveN * 64 + ni * 16 + ln;
            uint4 u0 = *(const uint4*)(&Bs[buf][r * 40 + hl * 8]);
            uint4 u1 = *(const uint4*)(&Bs[buf][r * 40 + 16 + hl * 8]);
            b[ni] = make_frag(u0, u1);
        }
        #pragma unroll
        for (int mi = 0; mi < 2; ++mi)
            #pragma unroll
            for (int ni = 0; ni < 4; ++ni)
                acc[mi][ni] = __builtin_amdgcn_wmma_f32_16x16x32_bf16(
                    false, a[mi], false, b[ni], (short)0, acc[mi][ni], false, false);
        __syncthreads();   // all waves done reading before next overwrite
        buf ^= 1;
    }

    // epilogue: bias, Q-scale, scatter to Qb / Kb / Vt(transposed)
    #pragma unroll
    for (int ni = 0; ni < 4; ++ni) {
        int n = nBase + waveN * 64 + ni * 16 + ln;
        int which = n >> 10;        // 0=Q 1=K 2=V
        int nn = n & (DN - 1);
        float bias = (which == 0) ? bq[nn] : (which == 1) ? bk[nn] : bv[nn];
        int h = nn >> 6, d = nn & 63;
        #pragma unroll
        for (int mi = 0; mi < 2; ++mi) {
            #pragma unroll
            for (int r = 0; r < 8; ++r) {
                int m = mBase + waveM * 32 + mi * 16 + r + hl * 8;
                int bi = m >> 11, s = m & (SN - 1);
                float val = acc[mi][ni][r] + bias;
                if (which == 0) {
                    val *= 0.125f;  // fold 1/sqrt(HD) into Q
                    Qb[((size_t)(bi * HN + h) * SN + s) * HDN + d] =
                        (unsigned short)f32_to_bf16(val);
                } else if (which == 1) {
                    Kb[((size_t)(bi * HN + h) * SN + s) * HDN + d] =
                        (unsigned short)f32_to_bf16(val);
                } else {
                    Vt[((size_t)(bi * HN + h) * HDN + d) * SN + s] =
                        (unsigned short)f32_to_bf16(val);
                }
            }
        }
    }
}

// ---------------------------------------------------------------------------
// Kernel 2: flash attention per (b,h).
// Grid (S/128, B*H); 8 waves/block; each wave owns 16 q-rows.
// K/V tiles are shared by all 8 waves: double-buffered async staging to LDS
// (K tile 32x64, V tile 64x32 per 32-k step), then ds_load fragment gathers.
// Per 32 k-positions: 4 WMMA for scores, online softmax, 4 WMMA for P*V.
// ---------------------------------------------------------------------------
__global__ __launch_bounds__(256) void attn_kernel(
    const unsigned short* __restrict__ Qb,  // [BH][S][64]
    const unsigned short* __restrict__ Kb,  // [BH][S][64]
    const unsigned short* __restrict__ Vt,  // [BH][64][S]
    const float* __restrict__ mask,         // [B][S]
    float* __restrict__ out) {              // [B][S][D]

    __shared__ unsigned short Ks[2][32 * 72];   // 32 rows x 64 d, stride 72
    __shared__ unsigned short Vs[2][64 * 40];   // 64 d-rows x 32 k, stride 40
    __shared__ unsigned short Ps[8 * 16 * 40];  // per-wave 16x32 bf16 P tile

    const int tid  = threadIdx.x;
    const int wave = tid >> 5;
    const int lane = tid & 31;
    const int ln   = lane & 15;
    const int hl   = lane >> 4;

    const int bh = blockIdx.y;
    const int bi = bh >> 4, h = bh & 15;
    const int qBase = blockIdx.x * 128 + wave * 16;

    const unsigned short* Qh = Qb + (size_t)bh * SN * HDN;
    const unsigned short* Kh = Kb + (size_t)bh * SN * HDN;
    const unsigned short* Vh = Vt + (size_t)bh * HDN * SN;
    const float* maskRow = mask + (size_t)bi * SN;
    const int wbase = wave * 16 * 40;

    // stage one K tile (32x64) + one V tile (64x32): 2 async ops per thread
    auto issue_kv = [&](int bufSel, int kb) {
        {
            int row = tid >> 3, cc = tid & 7;   // 32 rows x 8 chunks of 8 bf16
            async_copy_b128(Kh + (size_t)(kb + row) * HDN + cc * 8,
                            &Ks[bufSel][row * 72 + cc * 8]);
        }
        {
            int row = tid >> 2, cc = tid & 3;   // 64 rows x 4 chunks of 8 bf16
            async_copy_b128(Vh + (size_t)row * SN + kb + cc * 8,
                            &Vs[bufSel][row * 40 + cc * 8]);
        }
    };

    // Q fragments for this wave's 16 rows: d = 0..31 and 32..63
    const unsigned short* Qp = Qh + (size_t)(qBase + ln) * HDN;
    bfrag_t q0 = load_frag_row(Qp, hl);
    bfrag_t q1 = load_frag_row(Qp + 32, hl);

    v8f acc[4] = {};
    float mrun[8], lrun[8];
    #pragma unroll
    for (int r = 0; r < 8; ++r) { mrun[r] = -3.0e38f; lrun[r] = 0.0f; }

    issue_kv(0, 0);
    int buf = 0;
    for (int kb = 0; kb < SN; kb += 32) {
        bool has_next = (kb + 32) < SN;
        if (has_next) {
            issue_kv(buf ^ 1, kb + 32);
            asm volatile("s_wait_asynccnt 2" ::: "memory");
        } else {
            asm volatile("s_wait_asynccnt 0" ::: "memory");
        }
        __syncthreads();

        // ---- scores: 16 q-rows x 32 k-positions, pre-scaled by 0.125 via Q ----
        v8f c0 = {}, c1 = {};
        {
            const unsigned short* r0 = &Ks[buf][ln * 72];
            uint4 a0 = *(const uint4*)(r0 + hl * 8);
            uint4 a1 = *(const uint4*)(r0 + 16 + hl * 8);
            uint4 a2 = *(const uint4*)(r0 + 32 + hl * 8);
            uint4 a3 = *(const uint4*)(r0 + 48 + hl * 8);
            bfrag_t k0a = make_frag(a0, a1);
            bfrag_t k0b = make_frag(a2, a3);
            c0 = __builtin_amdgcn_wmma_f32_16x16x32_bf16(false, q0, false, k0a,
                                                         (short)0, c0, false, false);
            c0 = __builtin_amdgcn_wmma_f32_16x16x32_bf16(false, q1, false, k0b,
                                                         (short)0, c0, false, false);
            const unsigned short* r1 = &Ks[buf][(16 + ln) * 72];
            uint4 b0 = *(const uint4*)(r1 + hl * 8);
            uint4 b1 = *(const uint4*)(r1 + 16 + hl * 8);
            uint4 b2 = *(const uint4*)(r1 + 32 + hl * 8);
            uint4 b3 = *(const uint4*)(r1 + 48 + hl * 8);
            bfrag_t k1a = make_frag(b0, b1);
            bfrag_t k1b = make_frag(b2, b3);
            c1 = __builtin_amdgcn_wmma_f32_16x16x32_bf16(false, q0, false, k1a,
                                                         (short)0, c1, false, false);
            c1 = __builtin_amdgcn_wmma_f32_16x16x32_bf16(false, q1, false, k1b,
                                                         (short)0, c1, false, false);
        }
        // additive mask (broadcast over q-rows)
        float m0 = maskRow[kb + ln];
        float m1 = maskRow[kb + 16 + ln];
        #pragma unroll
        for (int r = 0; r < 8; ++r) { c0[r] += m0; c1[r] += m1; }

        // ---- online softmax (row = fixed m spans 16 lanes of one half) ----
        #pragma unroll
        for (int r = 0; r < 8; ++r) {
            float rm = fmaxf(c0[r], c1[r]);
            #pragma unroll
            for (int off = 8; off >= 1; off >>= 1)
                rm = fmaxf(rm, __shfl_xor(rm, off, 16));
            float mNew = fmaxf(mrun[r], rm);
            float alpha = __expf(mrun[r] - mNew);
            c0[r] = __expf(c0[r] - mNew);
            c1[r] = __expf(c1[r] - mNew);
            float rs = c0[r] + c1[r];
            #pragma unroll
            for (int off = 8; off >= 1; off >>= 1)
                rs += __shfl_xor(rs, off, 16);
            lrun[r] = lrun[r] * alpha + rs;
            mrun[r] = mNew;
            #pragma unroll
            for (int dt = 0; dt < 4; ++dt) acc[dt][r] *= alpha;
            // P -> per-wave LDS tile (bf16), row m = r + hl*8
            int m = r + hl * 8;
            Ps[wbase + m * 40 + ln]      = (unsigned short)f32_to_bf16(c0[r]);
            Ps[wbase + m * 40 + 16 + ln] = (unsigned short)f32_to_bf16(c1[r]);
        }

        // ---- P(16x32) @ V(32x64): A-frag + B-frags from LDS ----
        uint4 p0 = *(const uint4*)(&Ps[wbase + ln * 40 + hl * 8]);
        uint4 p1 = *(const uint4*)(&Ps[wbase + ln * 40 + 16 + hl * 8]);
        bfrag_t pf = make_frag(p0, p1);
        #pragma unroll
        for (int dt = 0; dt < 4; ++dt) {
            const unsigned short* vr = &Vs[buf][(dt * 16 + ln) * 40];
            uint4 u0 = *(const uint4*)(vr + hl * 8);
            uint4 u1 = *(const uint4*)(vr + 16 + hl * 8);
            bfrag_t vf = make_frag(u0, u1);
            acc[dt] = __builtin_amdgcn_wmma_f32_16x16x32_bf16(false, pf, false, vf,
                                                              (short)0, acc[dt],
                                                              false, false);
        }
        __syncthreads();   // all waves done with Ks/Vs[buf] before overwrite
        buf ^= 1;
    }

    // ---- normalize and write context: out[b, s, h*64+d] ----
    #pragma unroll
    for (int dt = 0; dt < 4; ++dt) {
        int d = dt * 16 + ln;
        #pragma unroll
        for (int r = 0; r < 8; ++r) {
            int m = r + hl * 8;
            int s = qBase + m;
            out[((size_t)(bi * SN + s)) * DN + h * HDN + d] = acc[dt][r] / lrun[r];
        }
    }
}

// ---------------------------------------------------------------------------
// Launch. d_in: 0=hidden[B,S,D] 1=mask[B,1,1,S] 2=Wq 3=bq 4=Wk 5=bk 6=Wv 7=bv
// d_out: [B,S,D] fp32. Workspace: Xb 16MB | Wb 6MB | Qb 16MB | Kb 16MB | Vt 16MB.
// ---------------------------------------------------------------------------
extern "C" void kernel_launch(void* const* d_in, const int* in_sizes, int n_in,
                              void* d_out, int out_size, void* d_ws, size_t ws_size,
                              hipStream_t stream) {
    const float* hs   = (const float*)d_in[0];
    const float* mask = (const float*)d_in[1];
    const float* Wq   = (const float*)d_in[2];
    const float* bq   = (const float*)d_in[3];
    const float* Wk   = (const float*)d_in[4];
    const float* bk   = (const float*)d_in[5];
    const float* Wv   = (const float*)d_in[6];
    const float* bv   = (const float*)d_in[7];
    float* out = (float*)d_out;

    char* ws = (char*)d_ws;
    unsigned short* Xb = (unsigned short*)ws;                       // 8192*1024
    unsigned short* Wb = Xb + (size_t)MROWS * DN;                   // 3072*1024
    unsigned short* Qb = Wb + (size_t)NQKV * DN;                    // 64*2048*64
    unsigned short* Kb = Qb + (size_t)BN * HN * SN * HDN;
    unsigned short* Vt = Kb + (size_t)BN * HN * SN * HDN;

    // 0) convert fp32 -> bf16
    {
        int n4 = (MROWS * DN) / 4;
        cvt_f32_bf16_kernel<<<(n4 + 255) / 256, 256, 0, stream>>>(hs, Xb, n4);
        int w4 = (DN * DN) / 4;
        cvt_f32_bf16_kernel<<<(w4 + 255) / 256, 256, 0, stream>>>(Wq, Wb, w4);
        cvt_f32_bf16_kernel<<<(w4 + 255) / 256, 256, 0, stream>>>(Wk, Wb + (size_t)DN * DN, w4);
        cvt_f32_bf16_kernel<<<(w4 + 255) / 256, 256, 0, stream>>>(Wv, Wb + (size_t)2 * DN * DN, w4);
    }
    // 1) fused QKV GEMM (async-LDS double buffered)
    {
        dim3 grid(MROWS / 128, NQKV / 128);
        qkv_gemm_kernel<<<grid, 256, 0, stream>>>(Xb, Wb, bq, bk, bv, Qb, Kb, Vt);
    }
    // 2) flash attention (async-LDS double buffered K/V)
    {
        dim3 grid(SN / 128, BN * HN);
        attn_kernel<<<grid, 256, 0, stream>>>(Qb, Kb, Vt, mask, out);
    }
    (void)in_sizes; (void)n_in; (void)out_size; (void)ws_size;
}
